// SimpleGCN_10780367913065
// MI455X (gfx1250) — matrix operand order repaired and checked by reading, not compiled, not run on hardware
//
#include <hip/hip_runtime.h>

#define NUM_GRAPHS 512
#define HDIM 64

typedef __attribute__((ext_vector_type(16))) __bf16 v16bf;
typedef __attribute__((ext_vector_type(8)))  float  v8f;

// ---------------- utility ----------------
__global__ void zero_f(float* __restrict__ p, int n) {
  int i = blockIdx.x * blockDim.x + threadIdx.x;
  if (i < n) p[i] = 0.0f;
}

// deg[dst[e]] += 1  (deg buffer pre-zeroed)
__global__ void degree_kernel(const int* __restrict__ dst, float* __restrict__ deg, int E) {
  int e = blockIdx.x * blockDim.x + threadIdx.x;
  if (e < E) unsafeAtomicAdd(&deg[dst[e]], 1.0f);
}

// deg -> dinv = rsqrt(deg + 1) in place
__global__ void dinv_kernel(float* __restrict__ deg, int N) {
  int i = blockIdx.x * blockDim.x + threadIdx.x;
  if (i < N) deg[i] = rsqrtf(deg[i] + 1.0f);
}

// ---------------- WMMA GEMM: H[N,64] = X[N,K] @ W[K,64]  (bf16 in, f32 acc) ----
// B fragments are pre-swizzled into LDS in the exact WMMA operand layout:
// record [(ks*4 + t)*32 + lane] = 16 contiguous bf16 (32B) -> one aligned
// v16bf LDS load per fragment in the hot loop (ds_load_b128 x2), instead of
// 16 ds_load_u16 + merges.
__global__ void __launch_bounds__(256)
gemm_wmma(const float* __restrict__ X, const float* __restrict__ W,
          float* __restrict__ H, int N, int K) {
  // max K=128 -> 4 ksteps * 4 tiles * 32 lanes * 16 bf16 = 16KB
  __shared__ __attribute__((aligned(32))) __bf16 ldsB[4 * 4 * 32 * 16];
  const int tid = threadIdx.x;
  const int nsteps = K >> 5;                 // K/32
  const int nentries = nsteps * 4 * 32;      // fragments * lanes

  // ---- stage + swizzle W into B-fragment layout ----
  for (int ent = tid; ent < nentries; ent += 256) {
    const int lane_e = ent & 31;
    const int t_e    = (ent >> 5) & 3;
    const int ks_e   = ent >> 7;
    const int kbase  = ks_e * 32 + (lane_e >> 4) * 8;    // B frag K rows
    const int col    = t_e * 16 + (lane_e & 15);         // output column
    __bf16* dstf = ldsB + (size_t)ent * 16;
    #pragma unroll
    for (int i = 0; i < 16; ++i)
      dstf[i] = (__bf16)W[(kbase + (i < 8 ? i : i + 8)) * HDIM + col];
  }
  __syncthreads();

  const int wave = tid >> 5;
  const int lane = tid & 31;
  const int m0 = (blockIdx.x * 8 + wave) * 16;
  if (m0 >= N) return;  // wave-uniform (N % 16 == 0)

  const int half  = lane >> 4;       // 0: lanes 0-15, 1: lanes 16-31
  const int mrow  = m0 + (lane & 15);
  const int kgrpA = half * 8;        // A frag K groups: [kgrpA..+7], [kgrpA+16..+23]
  const int ncol  = lane & 15;

  v8f c0 = {}, c1 = {}, c2 = {}, c3 = {};
  const float* xrow = X + (size_t)mrow * K;
  const v16bf* ldsBv = (const v16bf*)ldsB;

  for (int ks = 0; ks < nsteps; ++ks) {
    // ---- A fragment (16x32 bf16, ISA lane layout) ----
    const float* xr = xrow + ks * 32 + kgrpA;
    float4 p0 = *(const float4*)(xr);
    float4 p1 = *(const float4*)(xr + 4);
    float4 p2 = *(const float4*)(xr + 16);
    float4 p3 = *(const float4*)(xr + 20);
    v16bf a;
    a[0]=(__bf16)p0.x;  a[1]=(__bf16)p0.y;  a[2]=(__bf16)p0.z;  a[3]=(__bf16)p0.w;
    a[4]=(__bf16)p1.x;  a[5]=(__bf16)p1.y;  a[6]=(__bf16)p1.z;  a[7]=(__bf16)p1.w;
    a[8]=(__bf16)p2.x;  a[9]=(__bf16)p2.y;  a[10]=(__bf16)p2.z; a[11]=(__bf16)p2.w;
    a[12]=(__bf16)p3.x; a[13]=(__bf16)p3.y; a[14]=(__bf16)p3.z; a[15]=(__bf16)p3.w;

    // ---- B fragments: one aligned 32B LDS load each ----
    const v16bf* bf = ldsBv + (size_t)(ks * 4) * 32 + lane;
    v16bf b0 = bf[0 * 32];
    v16bf b1 = bf[1 * 32];
    v16bf b2 = bf[2 * 32];
    v16bf b3 = bf[3 * 32];

    c0 = __builtin_amdgcn_wmma_f32_16x16x32_bf16(false, a, false, b0, (short)0, c0, false, false);
    c1 = __builtin_amdgcn_wmma_f32_16x16x32_bf16(false, a, false, b1, (short)0, c1, false, false);
    c2 = __builtin_amdgcn_wmma_f32_16x16x32_bf16(false, a, false, b2, (short)0, c2, false, false);
    c3 = __builtin_amdgcn_wmma_f32_16x16x32_bf16(false, a, false, b3, (short)0, c3, false, false);
  }

  // ---- store D (f32 C/D layout: VGPR r -> M = r + 8*half, N = ncol) ----
  const int rowbase = m0 + half * 8;
  #pragma unroll
  for (int r = 0; r < 8; ++r) {
    float* hr = H + (size_t)(rowbase + r) * HDIM + ncol;
    hr[0]  = c0[r];
    hr[16] = c1[r];
    hr[32] = c2[r];
    hr[48] = c3[r];
  }
}

// ---------------- edge scatter-add: agg[dst] += h[src] * dinv[src]*dinv[dst] ----
// 16 threads per edge, float4 per thread; agg (25.6MB) is L2-resident ->
// global_atomic_add_f32 resolves in L2.
__global__ void edge_agg(const float* __restrict__ h, const int* __restrict__ src,
                         const int* __restrict__ dst, const float* __restrict__ dinv,
                         float* __restrict__ agg, int E) {
  int tid = blockIdx.x * blockDim.x + threadIdx.x;
  int e = tid >> 4;
  if (e >= E) return;
  int c = (tid & 15) * 4;
  int s = src[e], d = dst[e];
  float coef = dinv[s] * dinv[d];
  float4 v = *(const float4*)(h + (size_t)s * HDIM + c);
  float* out = agg + (size_t)d * HDIM + c;
  unsafeAtomicAdd(out + 0, v.x * coef);
  unsafeAtomicAdd(out + 1, v.y * coef);
  unsafeAtomicAdd(out + 2, v.z * coef);
  unsafeAtomicAdd(out + 3, v.w * coef);
}

// ---------------- h = relu(agg + h*dinv^2 + b), in place on h ----------------
__global__ void finalize_layer(float* __restrict__ h, const float* __restrict__ agg,
                               const float* __restrict__ dinv, const float* __restrict__ bias,
                               int N) {
  int tid = blockIdx.x * blockDim.x + threadIdx.x;
  int i = tid >> 4;
  if (i >= N) return;
  int c = (tid & 15) * 4;
  float di = dinv[i];
  float d2 = di * di;
  float4 hv = *(const float4*)(h + (size_t)i * HDIM + c);
  float4 av = *(const float4*)(agg + (size_t)i * HDIM + c);
  float4 bv = *(const float4*)(bias + c);
  float4 r;
  r.x = fmaxf(av.x + hv.x * d2 + bv.x, 0.0f);
  r.y = fmaxf(av.y + hv.y * d2 + bv.y, 0.0f);
  r.z = fmaxf(av.z + hv.z * d2 + bv.z, 0.0f);
  r.w = fmaxf(av.w + hv.w * d2 + bv.w, 0.0f);
  *(float4*)(h + (size_t)i * HDIM + c) = r;
}

// ---------------- pooling accumulation ----------------
__global__ void pool_accum(const float* __restrict__ h, const int* __restrict__ batch,
                           float* __restrict__ gsum, float* __restrict__ cnt, int N) {
  int tid = blockIdx.x * blockDim.x + threadIdx.x;
  int i = tid >> 4;
  if (i >= N) return;
  int c = (tid & 15) * 4;
  int g = batch[i];
  float4 v = *(const float4*)(h + (size_t)i * HDIM + c);
  float* out = gsum + (size_t)g * HDIM + c;
  unsafeAtomicAdd(out + 0, v.x);
  unsafeAtomicAdd(out + 1, v.y);
  unsafeAtomicAdd(out + 2, v.z);
  unsafeAtomicAdd(out + 3, v.w);
  if ((tid & 15) == 0) unsafeAtomicAdd(&cnt[g], 1.0f);
}

// ---------------- out[g,:] = (gsum[g]/max(cnt,1)) @ Wlin + blin --------------
__global__ void final_linear(const float* __restrict__ gsum, const float* __restrict__ cnt,
                             const float* __restrict__ Wlin, const float* __restrict__ blin,
                             float* __restrict__ out) {
  int g = blockIdx.x * blockDim.x + threadIdx.x;
  if (g >= NUM_GRAPHS) return;
  float invc = 1.0f / fmaxf(cnt[g], 1.0f);
  float acc0 = blin[0], acc1 = blin[1];
  #pragma unroll 8
  for (int k = 0; k < HDIM; ++k) {
    float m = gsum[(size_t)g * HDIM + k] * invc;
    acc0 += m * Wlin[k * 2 + 0];
    acc1 += m * Wlin[k * 2 + 1];
  }
  out[g * 2 + 0] = acc0;
  out[g * 2 + 1] = acc1;
}

extern "C" void kernel_launch(void* const* d_in, const int* in_sizes, int n_in,
                              void* d_out, int out_size, void* d_ws, size_t ws_size,
                              hipStream_t stream) {
  const float* x     = (const float*)d_in[0];
  const int*   edge  = (const int*)d_in[1];
  const int*   batch = (const int*)d_in[2];
  const float* W1    = (const float*)d_in[3];
  const float* b1    = (const float*)d_in[4];
  const float* W2    = (const float*)d_in[5];
  const float* b2    = (const float*)d_in[6];
  const float* Wlin  = (const float*)d_in[7];
  const float* blin  = (const float*)d_in[8];
  float* out = (float*)d_out;

  const int N = in_sizes[0] / 128;   // 100000
  const int E = in_sizes[1] / 2;     // 3200000
  const int* src = edge;
  const int* dst = edge + E;

  // workspace layout (floats)
  float* ws   = (float*)d_ws;
  float* hA   = ws;                    // [N,64]
  float* hB   = hA + (size_t)N * HDIM; // [N,64]
  float* dinv = hB + (size_t)N * HDIM; // [N]  (also deg accumulator)
  float* gsum = dinv + N;              // [512,64]
  float* cnt  = gsum + NUM_GRAPHS * HDIM; // [512]

  const int T = 256;
  const int nf    = N * HDIM;
  const int bNf   = (nf + T - 1) / T;
  const int bE    = (E + T - 1) / T;
  const int bE16  = (E * 16 + T - 1) / T;      // fits in int: 51.2M
  const int bN16  = (N * 16 + T - 1) / T;
  const int bN    = (N + T - 1) / T;
  const int bGemm = (N + 127) / 128;           // 8 waves x 16 rows per block

  // degrees -> dinv
  zero_f<<<bN, T, 0, stream>>>(dinv, N);
  degree_kernel<<<bE, T, 0, stream>>>(dst, dinv, E);
  dinv_kernel<<<bN, T, 0, stream>>>(dinv, N);

  // ---- layer 1 ----
  gemm_wmma<<<bGemm, T, 0, stream>>>(x, W1, hA, N, 128);
  zero_f<<<bNf, T, 0, stream>>>(hB, nf);
  edge_agg<<<bE16, T, 0, stream>>>(hA, src, dst, dinv, hB, E);
  finalize_layer<<<bN16, T, 0, stream>>>(hA, hB, dinv, b1, N);  // hA = relu(...)

  // ---- layer 2 ----
  gemm_wmma<<<bGemm, T, 0, stream>>>(hA, W2, hB, N, 64);        // hB = h2
  zero_f<<<bNf, T, 0, stream>>>(hA, nf);                        // hA = agg2
  edge_agg<<<bE16, T, 0, stream>>>(hB, src, dst, dinv, hA, E);
  finalize_layer<<<bN16, T, 0, stream>>>(hB, hA, dinv, b2, N);  // hB = relu(...)

  // ---- pooling + classifier ----
  zero_f<<<(NUM_GRAPHS * HDIM + T - 1) / T, T, 0, stream>>>(gsum, NUM_GRAPHS * HDIM);
  zero_f<<<(NUM_GRAPHS + T - 1) / T, T, 0, stream>>>(cnt, NUM_GRAPHS);
  pool_accum<<<bN16, T, 0, stream>>>(hB, batch, gsum, cnt, N);
  final_linear<<<(NUM_GRAPHS + T - 1) / T, T, 0, stream>>>(gsum, cnt, Wlin, blin, out);
}
